// Model_46660524703812
// MI455X (gfx1250) — matrix operand order repaired and checked by reading, not compiled, not run on hardware
//
#include <hip/hip_runtime.h>

typedef __attribute__((ext_vector_type(16))) _Float16 v16h;
typedef __attribute__((ext_vector_type(8)))  _Float16 v8h;
typedef __attribute__((ext_vector_type(8)))  float    v8f;

union Frag16 { v16h v; v8h h[2]; };

static __device__ __forceinline__ v8f wmma_f16(v16h a, v16h b, v8f c) {
  // D = A(16x32 f16) * B(32x16 f16) + C(16x16 f32)
  return __builtin_amdgcn_wmma_f32_16x16x32_f16(false, a, false, b, (short)0, c, false, false);
}

// ---------------- prep kernels ----------------

__global__ void k_cvt_f16(const float* __restrict__ in, _Float16* __restrict__ out, int n) {
  int i = blockIdx.x * 256 + threadIdx.x;
  if (i < n) out[i] = (_Float16)in[i];
}

// Wt[z][n][k] = W_z[k][n] as f16  (512x512 each)
__global__ void k_wt(const float* __restrict__ Wq, const float* __restrict__ Wk,
                     const float* __restrict__ Wv, const float* __restrict__ Wo,
                     _Float16* __restrict__ WtAll) {
  const int z = blockIdx.z;
  const float* W = z == 0 ? Wq : z == 1 ? Wk : z == 2 ? Wv : Wo;
  const int t = blockIdx.x * 256 + threadIdx.x;   // t = n*512 + k, 0..262143
  const int nn = t >> 9, kk = t & 511;
  WtAll[(size_t)z * 262144 + t] = (_Float16)W[kk * 512 + nn];
}

// P[s,r] = f*W1[0,r] + log(f+eps)*W1[1,r] + b1[r] ; Qf[s,r] = same w/o b1
__global__ void k_pq(const float* __restrict__ freqs, const float* __restrict__ W1,
                     const float* __restrict__ b1,
                     float* __restrict__ P, float* __restrict__ Qf) {
  const int t = blockIdx.x * 256 + threadIdx.x;   // 0 .. 4096*64-1
  const int sg = t >> 6, r = t & 63;
  const float f  = freqs[sg];
  const float lf = logf(f + 1e-6f);
  const float v  = f * W1[r] + lf * W1[64 + r];
  P[t]  = v + b1[r];
  Qf[t] = v;
}

// W2t[h][r] = W2[r][h]
__global__ void k_w2t(const float* __restrict__ W2, float* __restrict__ W2t) {
  const int idx = blockIdx.x * 256 + threadIdx.x;
  if (idx < 512) {
    const int h = idx >> 6, r = idx & 63;
    W2t[idx] = W2[r * 8 + h];
  }
}

// ---------------- WMMA GEMM: [4096x512] @ Wt^T + bias ----------------
// mode 0: Q16 [B,H,S,dh] (scaled 1/8)   mode 1: K16 [B,H,S,dh]
// mode 2: VT16 [B,H,dh,S]               mode 3: out32 [4096,512] fp32
__global__ __launch_bounds__(256) void k_gemm(
    const _Float16* __restrict__ x16, const _Float16* __restrict__ ctx16,
    const _Float16* __restrict__ WtAll,
    const float* __restrict__ bq, const float* __restrict__ bk,
    const float* __restrict__ bv, const float* __restrict__ bo,
    _Float16* __restrict__ Q16, _Float16* __restrict__ K16,
    _Float16* __restrict__ VT16, float* __restrict__ out32, int mode_base) {
  const int mode = mode_base + blockIdx.z;
  const int tid = threadIdx.x;
  const int wave = tid >> 5, lane = tid & 31;
  const int g = lane >> 4, nl = lane & 15;
  const int tile = blockIdx.x * 8 + wave;        // 0..8191
  const int r0 = (tile >> 5) * 16, n0 = (tile & 31) * 16;

  const _Float16* A  = (mode == 3) ? ctx16 : x16;
  const _Float16* Wt = WtAll + (size_t)mode * 262144;
  const float* bias  = mode == 0 ? bq : mode == 1 ? bk : mode == 2 ? bv : bo;

  const _Float16* arow = A  + (size_t)(r0 + nl) * 512;
  const _Float16* brow = Wt + (size_t)(n0 + nl) * 512;
  v8f c = {};
  for (int kk = 0; kk < 512; kk += 32) {
    Frag16 a, b;
    a.h[0] = *(const v8h*)(arow + kk + g * 8);          // A: K = g*8 .. g*8+7
    a.h[1] = *(const v8h*)(arow + kk + 16 + g * 8);     //    K = 16+g*8 ..
    b.h[0] = *(const v8h*)(brow + kk + g * 8);          // B: K = g*8 .. g*8+7
    b.h[1] = *(const v8h*)(brow + kk + 16 + g * 8);     //    K = 16+g*8 ..
    c = wmma_f16(a.v, b.v, c);
  }
  const float bval = bias[n0 + nl];
  const int n = n0 + nl, h = n >> 6, dcol = n & 63;
#pragma unroll
  for (int v = 0; v < 8; ++v) {
    const int m = v + 8 * g, r = r0 + m;
    const int bb = r >> 10, ss = r & 1023;
    const float val = c[v] + bval;
    if (mode == 0)
      Q16[(((size_t)(bb * 8 + h)) * 1024 + ss) * 64 + dcol] = (_Float16)(val * 0.125f);
    else if (mode == 1)
      K16[(((size_t)(bb * 8 + h)) * 1024 + ss) * 64 + dcol] = (_Float16)val;
    else if (mode == 2)
      VT16[(((size_t)(bb * 8 + h)) * 64 + dcol) * 1024 + ss] = (_Float16)val;
    else
      out32[(size_t)r * 512 + n] = val;
  }
}

// ---------------- fused flash attention + factored MLP bias ----------------
// grid (64 i-tiles, B). 8 waves = 8 heads. Q pre-scaled by 1/sqrt(dh).
__global__ __launch_bounds__(256) void k_attn(
    const _Float16* __restrict__ Q16, const _Float16* __restrict__ K16,
    const _Float16* __restrict__ VT16,
    const float* __restrict__ Pb, const float* __restrict__ Qfb,
    const float* __restrict__ W2t, const float* __restrict__ b2,
    _Float16* __restrict__ ctx16) {
  __shared__ float lP[16][64];          // P rows for this i-tile (b1 folded in)
  __shared__ float lQ[32][64];          // Qf rows for current j-tile
  __shared__ float lW2[8][64];          // W2^T
  __shared__ _Float16 lPr[8][16][32];   // per-wave prob staging (D->A reshape)

  const int it = blockIdx.x, b = blockIdx.y;
  const int tid = threadIdx.x;
  const int wave = tid >> 5, lane = tid & 31;
  const int g = lane >> 4, nl = lane & 15;
  const int i0 = it * 16;

  for (int idx = tid; idx < 512; idx += 256) lW2[idx >> 6][idx & 63] = W2t[idx];
  for (int idx = tid; idx < 1024; idx += 256)
    lP[idx >> 6][idx & 63] = Pb[((size_t)(b * 1024) + i0 + (idx >> 6)) * 64 + (idx & 63)];
  __syncthreads();

  Frag16 qa[2];
  const _Float16* qrow = Q16 + (((size_t)(b * 8 + wave)) * 1024 + i0 + nl) * 64;
  qa[0].h[0] = *(const v8h*)(qrow + g * 8);
  qa[0].h[1] = *(const v8h*)(qrow + 16 + g * 8);
  qa[1].h[0] = *(const v8h*)(qrow + 32 + g * 8);
  qa[1].h[1] = *(const v8h*)(qrow + 48 + g * 8);

  v8f o0 = {}, o1 = {}, o2 = {}, o3 = {};
  float Mrow[8], Lrow[8];
#pragma unroll
  for (int v = 0; v < 8; ++v) { Mrow[v] = -3e30f; Lrow[v] = 0.f; }
  const float b2h = b2[wave];
  const _Float16* kbase = K16 + ((size_t)(b * 8 + wave)) * 1024 * 64;
  const _Float16* vbase = VT16 + ((size_t)(b * 8 + wave)) * 64 * 1024;

  for (int j0 = 0; j0 < 1024; j0 += 32) {
    __syncthreads();
    for (int idx = tid; idx < 2048; idx += 256)
      lQ[idx >> 6][idx & 63] = Qfb[((size_t)(b * 1024) + j0 + (idx >> 6)) * 64 + (idx & 63)];
    __syncthreads();

    v8f s[2];
#pragma unroll
    for (int sub = 0; sub < 2; ++sub) {
      const _Float16* krow = kbase + (size_t)(j0 + sub * 16 + nl) * 64;
      Frag16 kb0, kb1;
      kb0.h[0] = *(const v8h*)(krow + g * 8);
      kb0.h[1] = *(const v8h*)(krow + 16 + g * 8);
      kb1.h[0] = *(const v8h*)(krow + 32 + g * 8);
      kb1.h[1] = *(const v8h*)(krow + 48 + g * 8);
      v8f cacc = {};
      cacc = wmma_f16(qa[0].v, kb0.v, cacc);      // dh 0..31
      cacc = wmma_f16(qa[1].v, kb1.v, cacc);      // dh 32..63
      // factored MLP bias: sum_r relu(P[i,r]-Qf[j,r]) * W2[r,h] + b2[h]
      float acc[8];
#pragma unroll
      for (int v = 0; v < 8; ++v) acc[v] = b2h;
      const int jl = sub * 16 + nl;
      for (int r = 0; r < 64; r += 4) {
        const float4 qv = *(const float4*)&lQ[jl][r];
        const float4 wv = *(const float4*)&lW2[wave][r];
#pragma unroll
        for (int v = 0; v < 8; ++v) {
          const float4 pv = *(const float4*)&lP[v + 8 * g][r];
          acc[v] += fmaxf(pv.x - qv.x, 0.f) * wv.x + fmaxf(pv.y - qv.y, 0.f) * wv.y +
                    fmaxf(pv.z - qv.z, 0.f) * wv.z + fmaxf(pv.w - qv.w, 0.f) * wv.w;
        }
      }
#pragma unroll
      for (int v = 0; v < 8; ++v) cacc[v] += acc[v];
      s[sub] = cacc;
    }

    // online softmax (row m = v + 8g lives in 16-lane half; reduce over n)
#pragma unroll
    for (int v = 0; v < 8; ++v) {
      float t = fmaxf(s[0][v], s[1][v]);
      t = fmaxf(t, __shfl_xor(t, 1, 32));
      t = fmaxf(t, __shfl_xor(t, 2, 32));
      t = fmaxf(t, __shfl_xor(t, 4, 32));
      t = fmaxf(t, __shfl_xor(t, 8, 32));
      const float mnew = fmaxf(Mrow[v], t);
      const float alpha = __expf(Mrow[v] - mnew);
      Mrow[v] = mnew;
      Lrow[v] *= alpha;
      o0[v] *= alpha; o1[v] *= alpha; o2[v] *= alpha; o3[v] *= alpha;
    }
#pragma unroll
    for (int sub = 0; sub < 2; ++sub) {
#pragma unroll
      for (int v = 0; v < 8; ++v) {
        const float p = __expf(s[sub][v] - Mrow[v]);
        lPr[wave][v + 8 * g][sub * 16 + nl] = (_Float16)p;
        float rs = p;
        rs += __shfl_xor(rs, 1, 32);
        rs += __shfl_xor(rs, 2, 32);
        rs += __shfl_xor(rs, 4, 32);
        rs += __shfl_xor(rs, 8, 32);
        Lrow[v] += rs;
      }
    }
    asm volatile("s_wait_dscnt 0" ::: "memory");   // probs staged in our own LDS slab
    Frag16 pa;                                     // A-layout: 16x32 probs
    pa.h[0] = *(const v8h*)&lPr[wave][nl][g * 8];
    pa.h[1] = *(const v8h*)&lPr[wave][nl][16 + g * 8];
#pragma unroll
    for (int dc = 0; dc < 4; ++dc) {
      const _Float16* vrow = vbase + (size_t)(dc * 16 + nl) * 1024 + j0;
      Frag16 vb;
      vb.h[0] = *(const v8h*)(vrow + g * 8);
      vb.h[1] = *(const v8h*)(vrow + 16 + g * 8);
      v8f& od = dc == 0 ? o0 : dc == 1 ? o1 : dc == 2 ? o2 : o3;
      od = wmma_f16(pa.v, vb.v, od);
    }
  }

#pragma unroll
  for (int v = 0; v < 8; ++v) {
    const float inv = 1.0f / Lrow[v];
    const int m = v + 8 * g;
    _Float16* crow = ctx16 + ((size_t)(b * 1024) + i0 + m) * 512 + wave * 64;
    crow[ 0 + nl] = (_Float16)(o0[v] * inv);
    crow[16 + nl] = (_Float16)(o1[v] * inv);
    crow[32 + nl] = (_Float16)(o2[v] * inv);
    crow[48 + nl] = (_Float16)(o3[v] * inv);
  }
}

// ---------------- launch ----------------
extern "C" void kernel_launch(void* const* d_in, const int* in_sizes, int n_in,
                              void* d_out, int out_size, void* d_ws, size_t ws_size,
                              hipStream_t stream) {
  const float* x     = (const float*)d_in[0];
  const float* freqs = (const float*)d_in[1];
  const float* Wq = (const float*)d_in[2];  const float* bq = (const float*)d_in[3];
  const float* Wk = (const float*)d_in[4];  const float* bk = (const float*)d_in[5];
  const float* Wv = (const float*)d_in[6];  const float* bv = (const float*)d_in[7];
  const float* Wo = (const float*)d_in[8];  const float* bo = (const float*)d_in[9];
  const float* W1 = (const float*)d_in[10]; const float* b1 = (const float*)d_in[11];
  const float* W2 = (const float*)d_in[12]; const float* b2 = (const float*)d_in[13];
  float* out = (float*)d_out;
  (void)in_sizes; (void)n_in; (void)out_size; (void)ws_size;

  char* ws = (char*)d_ws;                       // ~24 MB used
  _Float16* x16   = (_Float16*)(ws);
  _Float16* WtAll = (_Float16*)(ws + (4u  << 20));
  _Float16* Q16   = (_Float16*)(ws + (6u  << 20));
  _Float16* K16   = (_Float16*)(ws + (10u << 20));
  _Float16* VT16  = (_Float16*)(ws + (14u << 20));
  _Float16* ctx16 = (_Float16*)(ws + (18u << 20));
  float*    Pb    = (float*)   (ws + (22u << 20));
  float*    Qfb   = (float*)   (ws + (23u << 20));
  float*    W2t   = (float*)   (ws + (24u << 20));

  k_cvt_f16<<<8192, 256, 0, stream>>>(x, x16, 4096 * 512);
  k_wt<<<dim3(1024, 1, 4), 256, 0, stream>>>(Wq, Wk, Wv, Wo, WtAll);
  k_pq<<<1024, 256, 0, stream>>>(freqs, W1, b1, Pb, Qfb);
  k_w2t<<<2, 256, 0, stream>>>(W2, W2t);
  k_gemm<<<dim3(1024, 1, 3), 256, 0, stream>>>(x16, ctx16, WtAll, bq, bk, bv, bo,
                                               Q16, K16, VT16, out, 0);
  k_attn<<<dim3(64, 4), 256, 0, stream>>>(Q16, K16, VT16, Pb, Qfb, W2t, b2, ctx16);
  k_gemm<<<dim3(1024, 1, 1), 256, 0, stream>>>(x16, ctx16, WtAll, bq, bk, bv, bo,
                                               Q16, K16, VT16, out, 3);
}
